// GAT4_9294309228815
// MI455X (gfx1250) — matrix operand order repaired and checked by reading, not compiled, not run on hardware
//
#include <hip/hip_runtime.h>
#include <hip/hip_bf16.h>

typedef __attribute__((ext_vector_type(16))) _Float16 v16h;
typedef __attribute__((ext_vector_type(8)))  _Float16 v8h;
typedef __attribute__((ext_vector_type(8)))  float    v8f;

#define HC_ 128
#define NHEAD 4
#define CCH 32

// ---------- helpers ----------
__device__ __forceinline__ unsigned fkey(float f) {
    unsigned u = __float_as_uint(f);
    return (u & 0x80000000u) ? ~u : (u | 0x80000000u);
}
__device__ __forceinline__ float funkey(unsigned k) {
    unsigned u = (k & 0x80000000u) ? (k & 0x7FFFFFFFu) : ~k;
    return __uint_as_float(u);
}

// ---------- self-loop attr (mean of incoming edge_attr) ----------
__global__ void zero2_kernel(float* a, float* b, int n) {
    int i = blockIdx.x * 256 + threadIdx.x;
    if (i < n) { a[i] = 0.f; b[i] = 0.f; }
}
__global__ void loop_accum_kernel(const int* __restrict__ dst, const float* __restrict__ ea,
                                  float* cnt, float* sum, int E_) {
    int i = blockIdx.x * 256 + threadIdx.x;
    if (i >= E_) return;
    int d = dst[i];
    atomicAdd(&cnt[d], 1.0f);
    atomicAdd(&sum[d], ea[i]);
}
__global__ void loop_fin_kernel(float* sum, const float* cnt, int n) {
    int i = blockIdx.x * 256 + threadIdx.x;
    if (i >= n) return;
    float c = cnt[i];
    sum[i] = (c > 0.f) ? sum[i] / c : 0.f;   // in place: sum -> loop_attr
}

// ---------- layer-1 projection (K=7, VALU is fine) ----------
__global__ void gemm7_kernel(const float* __restrict__ x,
                             const float* __restrict__ Wl, const float* __restrict__ bl,
                             const float* __restrict__ Wr, const float* __restrict__ br,
                             float* __restrict__ xl, float* __restrict__ xr, int N_) {
    int i = blockIdx.x * 256 + threadIdx.x;
    if (i >= N_ * HC_) return;
    int n = i >> 7, f = i & 127;
    const float* row = x + n * 7;
    float al = bl[f], ar = br[f];
#pragma unroll
    for (int k = 0; k < 7; ++k) {
        float v = row[k];
        al += v * Wl[k * HC_ + f];
        ar += v * Wr[k * HC_ + f];
    }
    xl[i] = al; xr[i] = ar;
}

// ---------- split fp32 -> f16 hi/lo (with optional fused leaky_relu 0.01) ----------
__global__ void split_act_kernel(const float* __restrict__ h, _Float16* __restrict__ xh,
                                 _Float16* __restrict__ xlo, float slope, int total) {
    int i = blockIdx.x * 256 + threadIdx.x;
    if (i >= total) return;
    float v = h[i];
    v = (v > 0.f) ? v : slope * v;
    _Float16 hi = (_Float16)v;
    xh[i]  = hi;
    xlo[i] = (_Float16)(v - (float)hi);
}

// ---------- weight prep: fp32 [K=128][N=128] -> f16 hi/lo transposed [N][K] ----------
__global__ void prep_weight_kernel(const float* __restrict__ W,
                                   _Float16* __restrict__ WhT, _Float16* __restrict__ WlT) {
    int i = blockIdx.x * 256 + threadIdx.x;
    if (i >= HC_ * HC_) return;
    int k = i >> 7, n = i & 127;
    float w = W[i];
    _Float16 hi = (_Float16)w;
    WhT[n * HC_ + k] = hi;
    WlT[n * HC_ + k] = (_Float16)(w - (float)hi);
}

// ---------- WMMA GEMM: out[M x 128] = A(hi+lo) @ W(hi+lo) + bias ----------
// block = 256 threads (8 waves); block handles 16 rows; wave w -> column tile w.
// A fragment (16x32 f16): lane<16 row=lane&15, K in {kk+0..7, kk+16..23}; lane>=16 K+8.
// B fragment (32x16 f16): lane holds column (lane&15), K in {kk+0..7, kk+16..23} (+8 for lane>=16), same as A,
//   contiguous because W is pre-transposed.
__global__ void gemm_wmma_kernel(const _Float16* __restrict__ Ah, const _Float16* __restrict__ Al,
                                 const _Float16* __restrict__ BhT, const _Float16* __restrict__ BlT,
                                 const float* __restrict__ bias, float* __restrict__ out, int M) {
    const int wave = threadIdx.x >> 5;
    const int lane = threadIdx.x & 31;
    const int rowBase = blockIdx.x * 16;       // M % 16 == 0 (50000 = 3125*16)
    const int colBase = wave * 16;
    const int half = lane >> 4;
    const int l15  = lane & 15;
    const int arow = rowBase + l15;
    const int bcol = colBase + l15;
    v8f c = {};
#pragma unroll
    for (int kk = 0; kk < 128; kk += 32) {
        const int ak0 = kk + half * 8;
        const int ak1 = kk + 16 + half * 8;
        v8h a0h = *(const v8h*)(Ah + arow * HC_ + ak0);
        v8h a1h = *(const v8h*)(Ah + arow * HC_ + ak1);
        v8h a0l = *(const v8h*)(Al + arow * HC_ + ak0);
        v8h a1l = *(const v8h*)(Al + arow * HC_ + ak1);
        const int bk = kk + half * 8;
        v8h b0h = *(const v8h*)(BhT + bcol * HC_ + bk);
        v8h b1h = *(const v8h*)(BhT + bcol * HC_ + bk + 16);
        v8h b0l = *(const v8h*)(BlT + bcol * HC_ + bk);
        v8h b1l = *(const v8h*)(BlT + bcol * HC_ + bk + 16);
        v16h a_hi, a_lo, b_hi, b_lo;
#pragma unroll
        for (int j = 0; j < 8; ++j) {
            a_hi[j] = a0h[j]; a_hi[j + 8] = a1h[j];
            a_lo[j] = a0l[j]; a_lo[j + 8] = a1l[j];
            b_hi[j] = b0h[j]; b_hi[j + 8] = b1h[j];
            b_lo[j] = b0l[j]; b_lo[j + 8] = b1l[j];
        }
        c = __builtin_amdgcn_wmma_f32_16x16x32_f16(false, a_hi, false, b_hi, (short)0, c, false, false);
        c = __builtin_amdgcn_wmma_f32_16x16x32_f16(false, a_hi, false, b_lo, (short)0, c, false, false);
        c = __builtin_amdgcn_wmma_f32_16x16x32_f16(false, a_lo, false, b_hi, (short)0, c, false, false);
    }
    const int orow = rowBase + half * 8;
    const int ocol = colBase + l15;
    const float bv = bias[ocol];
#pragma unroll
    for (int i = 0; i < 8; ++i)
        out[(orow + i) * HC_ + ocol] = c[i] + bv;
}

// ---------- per-layer init: out = bias, mkey = 0 (== -inf key), den = 0 ----------
__global__ void init_layer_kernel(float* __restrict__ out, const float* __restrict__ bias,
                                  unsigned* __restrict__ mkey, float* __restrict__ den, int N_) {
    int i = blockIdx.x * 256 + threadIdx.x;
    if (i >= N_ * HC_) return;
    int f = i & 127;
    out[i] = bias[f];
    if (f < NHEAD) {
        int n = i >> 7;
        mkey[n * NHEAD + f] = 0u;
        den[n * NHEAD + f]  = 0.f;
    }
}

// ---------- edge pass 1: scores + segment max ----------
__global__ void edge_score_kernel(const int* __restrict__ src_idx, const int* __restrict__ dst_idx,
                                  const float* __restrict__ edge_attr, const float* __restrict__ loop_attr,
                                  const float* __restrict__ xl, const float* __restrict__ xr,
                                  const float* __restrict__ We, const float* __restrict__ att,
                                  float* __restrict__ score, unsigned* __restrict__ mkey,
                                  int EP, int Eorig) {
    int e = blockIdx.x * 8 + (threadIdx.x >> 5);
    if (e >= EP) return;
    int lane = threadIdx.x & 31;
    int s, d; float ea;
    if (e < Eorig) { s = src_idx[e]; d = dst_idx[e]; ea = edge_attr[e]; }
    else           { s = e - Eorig;  d = s;          ea = loop_attr[s]; }
    const float* xls = xl + (size_t)s * HC_;
    const float* xrd = xr + (size_t)d * HC_;
#pragma unroll
    for (int h = 0; h < NHEAD; ++h) {
        int idx = h * CCH + lane;
        float f = xls[idx] + xrd[idx] + ea * We[idx];
        f = (f > 0.f) ? f : 0.2f * f;               // GATv2 internal leaky_relu
        float p = f * att[idx];
#pragma unroll
        for (int off = 16; off; off >>= 1) p += __shfl_xor(p, off, 32);
        if (lane == 0) {
            score[(size_t)e * NHEAD + h] = p;
            atomicMax(&mkey[(size_t)d * NHEAD + h], fkey(p));
        }
    }
}

// ---------- edge pass 2: exp(score - max) (in place) + segment sum ----------
__global__ void edge_exp_kernel(const int* __restrict__ dst_idx, float* __restrict__ score,
                                const unsigned* __restrict__ mkey, float* __restrict__ den,
                                int EP, int Eorig) {
    int i = blockIdx.x * 256 + threadIdx.x;
    if (i >= EP * NHEAD) return;
    int e = i >> 2, h = i & 3;
    int d = (e < Eorig) ? dst_idx[e] : (e - Eorig);
    float m  = funkey(mkey[d * NHEAD + h]);
    float ex = __expf(score[i] - m);
    score[i] = ex;
    atomicAdd(&den[d * NHEAD + h], ex);
}

// ---------- edge pass 3: out[dst] += alpha * xl[src] ----------
__global__ void edge_aggr_kernel(const int* __restrict__ src_idx, const int* __restrict__ dst_idx,
                                 const float* __restrict__ ex, const float* __restrict__ den,
                                 const float* __restrict__ xl, float* __restrict__ out,
                                 int EP, int Eorig) {
    int e = blockIdx.x * 8 + (threadIdx.x >> 5);
    if (e >= EP) return;
    int lane = threadIdx.x & 31;
    int s, d;
    if (e < Eorig) { s = src_idx[e]; d = dst_idx[e]; }
    else           { s = e - Eorig;  d = s; }
#pragma unroll
    for (int h = 0; h < NHEAD; ++h) {
        float alpha = ex[(size_t)e * NHEAD + h] / den[(size_t)d * NHEAD + h];
        int idx = h * CCH + lane;
        atomicAdd(&out[(size_t)d * HC_ + idx], alpha * xl[(size_t)s * HC_ + idx]);
    }
}

// ---------- pooling + final linear ----------
__global__ void zero_kernel(float* p, int n) {
    int i = blockIdx.x * 256 + threadIdx.x;
    if (i < n) p[i] = 0.f;
}
__global__ void pool_kernel(const float* __restrict__ h, const int* __restrict__ batch,
                            float* __restrict__ pooled, float* __restrict__ gcnt, int N_) {
    int i = blockIdx.x * 256 + threadIdx.x;
    if (i >= N_ * HC_) return;
    int n = i >> 7, f = i & 127;
    int g = batch[n];
    atomicAdd(&pooled[g * HC_ + f], h[i]);
    if (f == 0) atomicAdd(&gcnt[g], 1.0f);
}
__global__ void final_lin_kernel(const float* __restrict__ pooled, const float* __restrict__ gcnt,
                                 const float* __restrict__ Wlin, const float* __restrict__ blin,
                                 float* __restrict__ out, int NG, int NCLS) {
    int i = threadIdx.x;
    if (i >= NG * NCLS) return;
    int g = i / NCLS, c = i % NCLS;
    float inv = 1.0f / fmaxf(gcnt[g], 1.0f);
    float acc = blin[c];
    for (int f = 0; f < HC_; ++f)
        acc += pooled[g * HC_ + f] * inv * Wlin[f * NCLS + c];
    out[i] = acc;
}

// ================= host side =================
extern "C" void kernel_launch(void* const* d_in, const int* in_sizes, int n_in,
                              void* d_out, int out_size, void* d_ws, size_t ws_size,
                              hipStream_t stream) {
    const float* x        = (const float*)d_in[0];
    const int*   eidx     = (const int*)  d_in[1];
    const float* eattr    = (const float*)d_in[2];
    const int*   batch    = (const int*)  d_in[3];
    const int E_ = in_sizes[1] / 2;
    const int N_ = in_sizes[3];
    const int EP = E_ + N_;
    const int* src = eidx;
    const int* dst = eidx + E_;

    const float *Wl[3], *bl[3], *Wr[3], *br[3], *We[3], *att[3], *bb[3];
    for (int L = 0; L < 3; ++L) {
        int base = 4 + L * 7;
        Wl[L]  = (const float*)d_in[base + 0];
        bl[L]  = (const float*)d_in[base + 1];
        Wr[L]  = (const float*)d_in[base + 2];
        br[L]  = (const float*)d_in[base + 3];
        We[L]  = (const float*)d_in[base + 4];
        att[L] = (const float*)d_in[base + 5];
        bb[L]  = (const float*)d_in[base + 6];
    }
    const float* Wlin = (const float*)d_in[25];
    const float* blin = (const float*)d_in[26];
    float* out = (float*)d_out;

    // bump allocator over workspace
    char* wp = (char*)d_ws;
    auto alloc = [&](size_t bytes) -> void* {
        void* p = (void*)wp;
        wp += (bytes + 255) & ~size_t(255);
        return p;
    };
    float*     cnt   = (float*)alloc(sizeof(float) * N_);
    float*     lattr = (float*)alloc(sizeof(float) * N_);            // ea_sum -> loop_attr
    float*     hA    = (float*)alloc(sizeof(float) * (size_t)N_ * HC_);
    float*     hB    = (float*)alloc(sizeof(float) * (size_t)N_ * HC_);
    float*     xlb   = (float*)alloc(sizeof(float) * (size_t)N_ * HC_);
    float*     xrb   = (float*)alloc(sizeof(float) * (size_t)N_ * HC_);
    _Float16*  xh    = (_Float16*)alloc(sizeof(_Float16) * (size_t)N_ * HC_);
    _Float16*  xlo   = (_Float16*)alloc(sizeof(_Float16) * (size_t)N_ * HC_);
    _Float16*  wlhT  = (_Float16*)alloc(sizeof(_Float16) * HC_ * HC_);
    _Float16*  wllT  = (_Float16*)alloc(sizeof(_Float16) * HC_ * HC_);
    _Float16*  wrhT  = (_Float16*)alloc(sizeof(_Float16) * HC_ * HC_);
    _Float16*  wrlT  = (_Float16*)alloc(sizeof(_Float16) * HC_ * HC_);
    float*     score = (float*)alloc(sizeof(float) * (size_t)EP * NHEAD);
    unsigned*  mkey  = (unsigned*)alloc(sizeof(unsigned) * (size_t)N_ * NHEAD);
    float*     den   = (float*)alloc(sizeof(float) * (size_t)N_ * NHEAD);
    float*     pooled= (float*)alloc(sizeof(float) * 8 * HC_);
    float*     gcnt  = (float*)alloc(sizeof(float) * 8);

    const int NF  = N_ * HC_;
    const int gNF = (NF + 255) / 256;
    const int gN  = (N_ + 255) / 256;
    const int gE  = (E_ + 255) / 256;
    const int gEP = (EP + 7) / 8;            // 8 edges (waves) per 256-thread block
    const int gEH = (EP * NHEAD + 255) / 256;
    const int gW  = (HC_ * HC_ + 255) / 256;
    const int gM  = N_ / 16;                 // N_ = 50000 is a multiple of 16

    // self-loop edge attr = per-target mean
    zero2_kernel<<<gN, 256, 0, stream>>>(cnt, lattr, N_);
    loop_accum_kernel<<<gE, 256, 0, stream>>>(dst, eattr, cnt, lattr, E_);
    loop_fin_kernel<<<gN, 256, 0, stream>>>(lattr, cnt, N_);

    auto edge_phase = [&](int L, float* outbuf) {
        init_layer_kernel<<<gNF, 256, 0, stream>>>(outbuf, bb[L], mkey, den, N_);
        edge_score_kernel<<<gEP, 256, 0, stream>>>(src, dst, eattr, lattr, xlb, xrb,
                                                   We[L], att[L], score, mkey, EP, E_);
        edge_exp_kernel<<<gEH, 256, 0, stream>>>(dst, score, mkey, den, EP, E_);
        edge_aggr_kernel<<<gEP, 256, 0, stream>>>(src, dst, score, den, xlb, outbuf, EP, E_);
    };

    // ---- layer 1 (K=7 projection on VALU) ----
    gemm7_kernel<<<gNF, 256, 0, stream>>>(x, Wl[0], bl[0], Wr[0], br[0], xlb, xrb, N_);
    edge_phase(0, hA);

    // ---- layer 2 (WMMA, split-f16; leaky_relu(0.01) fused into split) ----
    split_act_kernel<<<gNF, 256, 0, stream>>>(hA, xh, xlo, 0.01f, NF);
    prep_weight_kernel<<<gW, 256, 0, stream>>>(Wl[1], wlhT, wllT);
    prep_weight_kernel<<<gW, 256, 0, stream>>>(Wr[1], wrhT, wrlT);
    gemm_wmma_kernel<<<gM, 256, 0, stream>>>(xh, xlo, wlhT, wllT, bl[1], xlb, N_);
    gemm_wmma_kernel<<<gM, 256, 0, stream>>>(xh, xlo, wrhT, wrlT, br[1], xrb, N_);
    edge_phase(1, hB);

    // ---- layer 3 ----
    split_act_kernel<<<gNF, 256, 0, stream>>>(hB, xh, xlo, 0.01f, NF);
    prep_weight_kernel<<<gW, 256, 0, stream>>>(Wl[2], wlhT, wllT);
    prep_weight_kernel<<<gW, 256, 0, stream>>>(Wr[2], wrhT, wrlT);
    gemm_wmma_kernel<<<gM, 256, 0, stream>>>(xh, xlo, wlhT, wllT, bl[2], xlb, N_);
    gemm_wmma_kernel<<<gM, 256, 0, stream>>>(xh, xlo, wrhT, wrlT, br[2], xrb, N_);
    edge_phase(2, hA);   // no activation after layer 3

    // ---- mean pool + classifier ----
    zero_kernel<<<1, 256, 0, stream>>>(pooled, 8 * HC_);
    zero_kernel<<<1, 256, 0, stream>>>(gcnt, 8);
    pool_kernel<<<gNF, 256, 0, stream>>>(hA, batch, pooled, gcnt, N_);
    final_lin_kernel<<<1, 64, 0, stream>>>(pooled, gcnt, Wlin, blin, out, 8, 5);
}
